// MultiHeadedGAT_8735963480440
// MI455X (gfx1250) — matrix operand order repaired and checked by reading, not compiled, not run on hardware
//
#include <hip/hip_runtime.h>

#define B_ 4
#define P_ 2048
#define C_ 512
#define D_ 1024
#define H_ 16
#define DH_ 64

typedef __bf16 v16bf __attribute__((ext_vector_type(16)));
typedef float v8f __attribute__((ext_vector_type(8)));
typedef unsigned int v4u __attribute__((ext_vector_type(4)));
typedef int v8i_t __attribute__((ext_vector_type(8)));
typedef int v4i_t __attribute__((ext_vector_type(4)));

union Frag16 {
  v16bf v;
  uint4 u[2];
  unsigned short s[16];
};

__device__ __forceinline__ unsigned short f2bf(float f) {
  unsigned int u = __float_as_uint(f);
  u += 0x7fffu + ((u >> 16) & 1u);   // round-to-nearest-even
  return (unsigned short)(u >> 16);
}

__device__ __forceinline__ v8f wmma_bf16(const Frag16& a, const Frag16& b, v8f c) {
  return __builtin_amdgcn_wmma_f32_16x16x32_bf16(
      /*neg_a=*/false, a.v, /*neg_b=*/false, b.v,
      /*c_mod=*/(short)0, c, /*reuse_a=*/false, /*reuse_b=*/false);
}

// --------------------------------------------------------------- TDM tile load
// 2-D tile via Tensor Data Mover: 16 rows x 1024 bf16 elements, row stride 1024
// elements, global -> LDS. Descriptor packing per CDNA5 ISA.
__device__ __forceinline__ void tdm_load_wtile(const unsigned short* gsrc,
                                               unsigned int lds_off) {
  const unsigned long long ga = (unsigned long long)(uintptr_t)gsrc;
  v4u g0 = {0u, 0u, 0u, 0u};
  g0.x = 1u;                                          // count=1, user D#
  g0.y = lds_off;                                     // lds_addr (bytes)
  g0.z = (unsigned int)(ga & 0xffffffffu);            // global_addr[31:0]
  g0.w = (unsigned int)((ga >> 32) & 0x01ffffffu)     // global_addr[56:32]
         | (2u << 30);                                // type=2 ("image")
  v8i_t g1 = {
      (int)(1u << 16),        // w0: wg_mask=0, data_size=1 (2 bytes)
      (int)(1024u << 16),     // w1: atomic_addr=0, tensor_dim0 lo16 = 1024
      (int)(1024u << 16),     // w2: tensor_dim0 hi=0, tensor_dim1 lo16 = 1024
      (int)(1024u << 16),     // w3: tensor_dim1 hi=0, tile_dim0 = 1024
      16,                     // w4: tile_dim1=16, tile_dim2=0
      1024,                   // w5: tensor_dim0_stride lo32 = 1024
      0,                      // w6: stride0 hi=0, tensor_dim1_stride lo16=0
      0};                     // w7
  v4i_t gz = {0, 0, 0, 0};
#if __clang_major__ >= 23
  v8i_t gz8 = {0, 0, 0, 0, 0, 0, 0, 0};
  __builtin_amdgcn_tensor_load_to_lds(g0, g1, gz, gz, gz8, 0);
#else
  __builtin_amdgcn_tensor_load_to_lds(g0, g1, gz, gz, 0);
#endif
}

// ---------------------------------------------------------------- cast f32->bf16
__global__ void cast_bf16_kernel(const float* __restrict__ src,
                                 unsigned short* __restrict__ dst, long n) {
  long i = (long)blockIdx.x * blockDim.x + threadIdx.x;
  long stride = (long)gridDim.x * blockDim.x;
  for (; i < n; i += stride) dst[i] = f2bf(src[i]);
}

// ------------------------------------------------- transpose + cast weights (DxD)
// Wt[n][k] = W[k][n], bf16 output, so GEMM B-fragments load contiguously in k.
__global__ void transpose_cast_kernel(const float* __restrict__ W,
                                      unsigned short* __restrict__ Wt) {
  __shared__ float tile[32][33];
  const int bx = blockIdx.x * 32, by = blockIdx.y * 32;
  const int tx = threadIdx.x & 31, ty = threadIdx.x >> 5;  // 256 thr: ty 0..7
  for (int i = ty; i < 32; i += 8)
    tile[i][tx] = W[(long)(by + i) * D_ + bx + tx];
  __syncthreads();
  for (int i = ty; i < 32; i += 8)
    Wt[(long)(bx + i) * D_ + by + tx] = f2bf(tile[tx][i]);
}

// ------------------------------------------------------------ projection GEMM
// Out[M,D] = bf16( (A[M,D] @ W[D,D] + bias) * scale ),  W given as Wt[n][k].
// One wave = one 16x16 output tile, K-loop of 32. 8 waves per block (8 m-tiles)
// sharing one 16-column Wt tile staged into LDS by the Tensor Data Mover.
// If doT, also writes batch-transposed copy OutT[b][d][c] (for V), C_=512 rows.
__global__ void gemm_proj_kernel(const unsigned short* __restrict__ A,
                                 const unsigned short* __restrict__ Wt,
                                 const float* __restrict__ bias,
                                 unsigned short* __restrict__ Out,
                                 unsigned short* __restrict__ OutT,
                                 int M, float scale, int doT) {
  __shared__ __align__(16) unsigned short sB[16][D_];   // 32 KB weight tile
  const int lane = threadIdx.x & 31;
  const int wave = threadIdx.x >> 5;
  const int row0 = (blockIdx.y * 8 + wave) * 16;
  const int col0 = blockIdx.x * 16;

  if (wave == 0) {
    tdm_load_wtile(Wt + (size_t)col0 * D_,
                   (unsigned int)(uintptr_t)&sB[0][0]);
    __builtin_amdgcn_s_wait_tensorcnt(0);
  }
  __syncthreads();

  const int nA = lane & 15;                // A row / B col / D col within tile
  const int hiA = (lane >> 4) * 8;         // A frag K-group offset
  const int hiB = (lane >> 4) * 8;         // B frag K-group offset
  const uint4* A4 = (const uint4*)A;
  const uint4* B4 = (const uint4*)&sB[nA][0];
  const long arow = (long)(row0 + nA) * D_;
  v8f acc = {};
  for (int kk = 0; kk < D_; kk += 32) {
    Frag16 af, bf;
    const long ka = arow + kk + hiA;
    af.u[0] = A4[ka >> 3];
    af.u[1] = A4[(ka + 16) >> 3];
    const int kb = (kk + hiB) >> 3;
    bf.u[0] = B4[kb];
    bf.u[1] = B4[kb + 2];
    acc = wmma_bf16(af, bf, acc);
  }
  const float bv = bias[col0 + nA];
  const int mhi = (lane >> 4) * 8;
#pragma unroll
  for (int r = 0; r < 8; ++r) {
    const int m = r + mhi;
    const unsigned short o = f2bf((acc[r] + bv) * scale);
    Out[(long)(row0 + m) * D_ + col0 + nA] = o;
    if (doT) {
      const int grow = row0 + m;
      const int b = grow >> 9;            // / C_ (512)
      const int c = grow & (C_ - 1);      // % C_
      OutT[((long)b * D_ + col0 + nA) * C_ + c] = o;
    }
  }
}

// ------------------------------------------------------------- fused attention
// One block = one (b, h, 16-row P tile). 8 waves.
//  phase 1: scores[16][512] = q . k^T (WMMA, each wave 4 c-tiles)
//  phase 2: masked softmax + edge weighting (f32, 16 thr/row) -> bf16 in LDS
//  phase 3: ctx[16][64] = attn @ v   (WMMA from LDS A-frags, vT B-frags)
__global__ void attention_kernel(const unsigned short* __restrict__ q,
                                 const unsigned short* __restrict__ k,
                                 const unsigned short* __restrict__ vT,
                                 const float* __restrict__ edge,
                                 float* __restrict__ ctx) {
  __shared__ __align__(16) float sc[16][C_];            // 32 KB scores
  __shared__ __align__(16) unsigned short at[16][C_];   // 16 KB attn bf16
  __shared__ float red[16][16];
  __shared__ float rowmax[16];
  __shared__ float rowsum[16];
  __shared__ float part[8][16][16];                     // 8 KB partial ctx

  const int b = blockIdx.z, h = blockIdx.y, pt = blockIdx.x;
  const int lane = threadIdx.x & 31;
  const int wave = threadIdx.x >> 5;
  const int mA = lane & 15;
  const int hiA = (lane >> 4) * 8;
  const int hiB = (lane >> 4) * 8;
  const int p0 = pt * 16;

  const uint4* q4 = (const uint4*)q;
  const uint4* k4 = (const uint4*)k;
  const uint4* v4 = (const uint4*)vT;

  // ---- phase 1: scores = q @ k^T (per head, DH=64 = two 32-K WMMA steps)
  const long qrow = ((long)(b * P_ + p0 + mA)) * D_ + h * DH_;
  for (int ct = wave * 4; ct < wave * 4 + 4; ++ct) {
    v8f acc = {};
    const long krow = ((long)(b * C_ + ct * 16 + mA)) * D_ + h * DH_;
#pragma unroll
    for (int kk = 0; kk < DH_; kk += 32) {
      Frag16 af, bf;
      const long ka = qrow + kk + hiA;
      af.u[0] = q4[ka >> 3];
      af.u[1] = q4[(ka + 16) >> 3];
      const long kb = krow + kk + hiB;
      bf.u[0] = k4[kb >> 3];
      bf.u[1] = k4[(kb + 16) >> 3];
      acc = wmma_bf16(af, bf, acc);
    }
    const int mhi = (lane >> 4) * 8;
#pragma unroll
    for (int r = 0; r < 8; ++r) sc[r + mhi][ct * 16 + mA] = acc[r];
  }
  __syncthreads();

  // ---- phase 2: masked softmax, 16 threads per row
  const int row = threadIdx.x >> 4;
  const int tcol = threadIdx.x & 15;
  const float* erow = edge + ((long)b * P_ + p0 + row) * C_;
  float mx = -3.0e38f;
  for (int c = tcol; c < C_; c += 16) {
    const float e = erow[c];
    const float s = (e > 0.f) ? sc[row][c] : -1.0e18f;
    sc[row][c] = s;
    mx = fmaxf(mx, s);
  }
  red[row][tcol] = mx;
  __syncthreads();
  if (tcol == 0) {
    float m2 = -3.0e38f;
    for (int i = 0; i < 16; ++i) m2 = fmaxf(m2, red[row][i]);
    rowmax[row] = m2;
  }
  __syncthreads();
  const float mr = rowmax[row];
  float sum = 0.f;
  for (int c = tcol; c < C_; c += 16) {
    const float ex = __expf(sc[row][c] - mr);
    sc[row][c] = ex;
    sum += ex;
  }
  red[row][tcol] = sum;
  __syncthreads();
  if (tcol == 0) {
    float s2 = 0.f;
    for (int i = 0; i < 16; ++i) s2 += red[row][i];
    rowsum[row] = s2;
  }
  __syncthreads();
  const float inv = 1.0f / rowsum[row];
  for (int c = tcol; c < C_; c += 16)
    at[row][c] = f2bf(sc[row][c] * inv * erow[c]);
  __syncthreads();

  // ---- phase 3: ctx = attn @ v ; wave -> (ntile = w&3, K half = w>>2)
  const int ntile = wave & 3;
  const int khalf = wave >> 2;
  v8f acc = {};
  const long vrow = ((long)(b * D_) + h * DH_ + ntile * 16 + mA) * C_;
  const uint4* l4 = (const uint4*)&at[mA][0];
  for (int kk = khalf * 256; kk < khalf * 256 + 256; kk += 32) {
    Frag16 af, bf;
    const int ka = kk + hiA;
    af.u[0] = l4[ka >> 3];
    af.u[1] = l4[(ka + 16) >> 3];
    const long kb = vrow + kk + hiB;
    bf.u[0] = v4[kb >> 3];
    bf.u[1] = v4[(kb + 16) >> 3];
    acc = wmma_bf16(af, bf, acc);
  }
  const int mhi = (lane >> 4) * 8;
#pragma unroll
  for (int r = 0; r < 8; ++r) part[wave][r + mhi][mA] = acc[r];
  __syncthreads();

  // ---- phase 4: reduce K halves, write ctx
  for (int idx = threadIdx.x; idx < 4 * 16 * 16; idx += 256) {
    const int nt = idx >> 8;
    const int m = (idx >> 4) & 15;
    const int n = idx & 15;
    const float v = part[nt][m][n] + part[nt + 4][m][n];
    ctx[((long)(b * P_ + p0 + m)) * D_ + h * DH_ + nt * 16 + n] = v;
  }
}

// ---------------------------------------------------------------- LayerNorm(D)
__global__ void layernorm_kernel(const float* __restrict__ ctx,
                                 const float* __restrict__ gamma,
                                 const float* __restrict__ beta,
                                 float* __restrict__ out) {
  __shared__ float s1[256];
  __shared__ float s2[256];
  const long row = blockIdx.x;
  const float* x = ctx + row * D_;
  const int t = threadIdx.x;
  float a = 0.f, b = 0.f;
  for (int i = t; i < D_; i += 256) {
    const float v = x[i];
    a += v;
    b += v * v;
  }
  s1[t] = a;
  s2[t] = b;
  __syncthreads();
  for (int st = 128; st > 0; st >>= 1) {
    if (t < st) {
      s1[t] += s1[t + st];
      s2[t] += s2[t + st];
    }
    __syncthreads();
  }
  const float mean = s1[0] * (1.0f / D_);
  const float var = s2[0] * (1.0f / D_) - mean * mean;
  const float inv = rsqrtf(var + 1e-6f);
  for (int i = t; i < D_; i += 256)
    out[row * D_ + i] = (x[i] - mean) * inv * gamma[i] + beta[i];
}

// ------------------------------------------------------------------- launcher
extern "C" void kernel_launch(void* const* d_in, const int* in_sizes, int n_in,
                              void* d_out, int out_size, void* d_ws, size_t ws_size,
                              hipStream_t stream) {
  (void)in_sizes; (void)n_in; (void)out_size; (void)ws_size;
  const float* para    = (const float*)d_in[0];
  const float* cluster = (const float*)d_in[1];
  const float* edge    = (const float*)d_in[2];
  const float* Wq      = (const float*)d_in[3];
  const float* bq      = (const float*)d_in[4];
  const float* Wk      = (const float*)d_in[5];
  const float* bk      = (const float*)d_in[6];
  const float* Wv      = (const float*)d_in[7];
  const float* bv      = (const float*)d_in[8];
  const float* gamma   = (const float*)d_in[9];
  const float* beta    = (const float*)d_in[10];
  float* out = (float*)d_out;

  char* ws = (char*)d_ws;
  size_t off = 0;
  auto take = [&](size_t bytes) -> char* {
    char* p = ws + off;
    off += (bytes + 255) & ~(size_t)255;
    return p;
  };
  unsigned short* para_bf = (unsigned short*)take((size_t)B_ * P_ * D_ * 2);
  unsigned short* clus_bf = (unsigned short*)take((size_t)B_ * C_ * D_ * 2);
  unsigned short* wqt     = (unsigned short*)take((size_t)D_ * D_ * 2);
  unsigned short* wkt     = (unsigned short*)take((size_t)D_ * D_ * 2);
  unsigned short* wvt     = (unsigned short*)take((size_t)D_ * D_ * 2);
  unsigned short* qb      = (unsigned short*)take((size_t)B_ * P_ * D_ * 2);
  unsigned short* kb      = (unsigned short*)take((size_t)B_ * C_ * D_ * 2);
  unsigned short* vb      = (unsigned short*)take((size_t)B_ * C_ * D_ * 2);
  unsigned short* vT      = (unsigned short*)take((size_t)B_ * D_ * C_ * 2);
  float* ctx              = (float*)take((size_t)B_ * P_ * D_ * 4);

  cast_bf16_kernel<<<512, 256, 0, stream>>>(para, para_bf, (long)B_ * P_ * D_);
  cast_bf16_kernel<<<256, 256, 0, stream>>>(cluster, clus_bf, (long)B_ * C_ * D_);

  dim3 tg(D_ / 32, D_ / 32);
  transpose_cast_kernel<<<tg, 256, 0, stream>>>(Wq, wqt);
  transpose_cast_kernel<<<tg, 256, 0, stream>>>(Wk, wkt);
  transpose_cast_kernel<<<tg, 256, 0, stream>>>(Wv, wvt);

  // Q = (para@Wq + bq) * (1/sqrt(DH))
  gemm_proj_kernel<<<dim3(D_ / 16, (B_ * P_) / 128), 256, 0, stream>>>(
      para_bf, wqt, bq, qb, nullptr, B_ * P_, 0.125f, 0);
  // K = cluster@Wk + bk
  gemm_proj_kernel<<<dim3(D_ / 16, (B_ * C_) / 128), 256, 0, stream>>>(
      clus_bf, wkt, bk, kb, nullptr, B_ * C_, 1.0f, 0);
  // V = cluster@Wv + bv   (also batch-transposed vT[b][d][c])
  gemm_proj_kernel<<<dim3(D_ / 16, (B_ * C_) / 128), 256, 0, stream>>>(
      clus_bf, wvt, bv, vb, vT, B_ * C_, 1.0f, 1);

  attention_kernel<<<dim3(P_ / 16, H_, B_), 256, 0, stream>>>(qb, kb, vT, edge, ctx);
  layernorm_kernel<<<B_ * P_, 256, 0, stream>>>(ctx, gamma, beta, out);
}
